// RIMs_87952340287561
// MI455X (gfx1250) — matrix operand order repaired and checked by reading, not compiled, not run on hardware
//
#include <hip/hip_runtime.h>

// RIMs forward for MI455X (gfx1250, wave32, WMMA bf16 -> f32 accumulate).
//
// Pipeline (5 kernels, one stream, stream order = dependency order):
//  1) rims_prep_weights : f32 weights -> bf16 WMMA-B-swizzled 1KB tiles
//  2) rims_zero_state   : zero h (f32 + padded bf16 copies)
//  3) rims_xproj        : KX = x@Wk (f32), VX = x@Wv (bf16); 1Mx4N / 1Mx5N blocked
//  4) rims_uproj        : UX[n] = VX@W_ih[n] (bf16); 2Mx5N register-blocked
//  5) rims_recurrence   : ONE launch, 16 WGs x 512 thr; each WG owns a 16-batch
//     tile for all 128 steps (reference has no cross-batch coupling).
//     3 workgroup barriers per step; s = q.kx reduced from WMMA accumulators
//     via shfl_xor; all-rim gh staged in bf16 LDS.
//
// Workspace requirement: ~151 MiB.

typedef __attribute__((ext_vector_type(16))) __bf16 v16bf;
typedef __attribute__((ext_vector_type(8)))  __bf16 v8bf;
typedef __attribute__((ext_vector_type(8)))  float  v8f;

#define T_STEPS 128
#define BATCH   256
#define N_RIMS  6
#define K_ACT   4
#define D_HID   100
#define G3      300
#define TBROWS  (T_STEPS*BATCH)

// padded K-tile / N-tile counts
#define KT_DIN 19   // 608/32  (D_IN=600)
#define KT_H   4    // 128/32  (D_HID=100)
#define KT_DV  13   // 416/32  (D_V=400)
#define NT_DK  4    // 64/16
#define NT_DV  25   // 400/16
#define NT_G3  19   // 304/16  (3*D_HID=300)

// ws layout (bytes); every region is 1KB-aligned
#define OFF_WK   ((size_t)0)
#define OFF_WV   (OFF_WK  + (size_t)KT_DIN*NT_DK*1024)            // 76 tiles
#define OFF_WQ   (OFF_WV  + (size_t)KT_DIN*NT_DV*1024)            // 475 tiles
#define OFF_WHH  (OFF_WQ  + (size_t)N_RIMS*KT_H*NT_DK*1024)       // 96 tiles
#define OFF_WIH  (OFF_WHH + (size_t)N_RIMS*KT_H*NT_G3*1024)       // 456 tiles
#define OFF_KX   (OFF_WIH + (size_t)N_RIMS*KT_DV*NT_G3*1024)      // 1482 tiles
#define OFF_VX   (OFF_KX  + (size_t)TBROWS*64*4)                  // f32 [TB,64]
#define OFF_UX   (OFF_VX  + (size_t)TBROWS*400*2)                 // bf16 [TB,400]
#define OFF_HF   (OFF_UX  + (size_t)TBROWS*6*304*2)               // bf16 [TB,6,304]
#define OFF_HBF  (OFF_HF  + (size_t)BATCH*6*100*4)                // f32 h
// end = OFF_HBF + 6*256*128*2  (~157.8 MB)                       // bf16 h, K padded 128

__device__ __forceinline__ v8bf bz8() {
  union { unsigned u[4]; v8bf h; } z;
  z.u[0] = 0u; z.u[1] = 0u; z.u[2] = 0u; z.u[3] = 0u;
  return z.h;
}

__device__ __forceinline__ v8f wmma_bf16(v16bf a, v16bf b, v8f c) {
  return __builtin_amdgcn_wmma_f32_16x16x32_bf16(false, a, false, b, (short)0, c, false, false);
}

// A-fragment (16-bit 16x32 layout): lane<16 -> row = lane, K in {kb..kb+7} u
// {kb+16..kb+23} with kb = kt*32; lanes 16..31 same rows, kb += 8.
__device__ __forceinline__ v16bf load_a_bf16(const __bf16* rowp, int kb) {
  union { v16bf v; v8bf h[2]; } a;
  a.h[0] = *(const v8bf*)(rowp + kb);
  a.h[1] = *(const v8bf*)(rowp + kb + 16);
  return a.v;
}

// ---------------------------------------------------------------------------
// Kernel 1: weights -> bf16 WMMA-B-swizzled tiles (32x16 KxN per 1KB tile):
// lane<16 -> col=lane, K=kt*32+{0..7,16..23}; lane>=16 -> col=lane-16, K=kt*32+{8..15,24..31}.
// ---------------------------------------------------------------------------
__global__ void rims_prep_weights(const float* __restrict__ Wq, const float* __restrict__ Wk,
                                  const float* __restrict__ Wv, const float* __restrict__ Wih,
                                  const float* __restrict__ Whh, char* __restrict__ ws) {
  const int bid = blockIdx.x;
  const int l = threadIdx.x;  // 0..31
  const float* src; int srcK, srcC, ldc, kt, nt; __bf16* dst;
  if (bid < 76) {                                  // Wk [600,64], KT=19 NT=4
    int tt = bid; nt = tt/19; kt = tt - nt*19;
    src = Wk; srcK = 600; srcC = 64; ldc = 64;
    dst = (__bf16*)(ws + OFF_WK) + (size_t)tt*512;
  } else if (bid < 551) {                          // Wv [600,400], KT=19 NT=25
    int tt = bid - 76; nt = tt/19; kt = tt - nt*19;
    src = Wv; srcK = 600; srcC = 400; ldc = 400;
    dst = (__bf16*)(ws + OFF_WV) + (size_t)tt*512;
  } else if (bid < 647) {                          // Wq [6][100,64], KT=4 NT=4
    int tt = bid - 551; int n = tt >> 4; int rr = tt & 15; nt = rr >> 2; kt = rr & 3;
    src = Wq + (size_t)n*100*64; srcK = 100; srcC = 64; ldc = 64;
    dst = (__bf16*)(ws + OFF_WQ) + (size_t)tt*512;
  } else if (bid < 1103) {                         // W_hh [6][100,300], KT=4 NT=19
    int tt = bid - 647; int n = tt/76; int rr = tt - n*76; nt = rr >> 2; kt = rr & 3;
    src = Whh + (size_t)n*100*300; srcK = 100; srcC = 300; ldc = 300;
    dst = (__bf16*)(ws + OFF_WHH) + (size_t)tt*512;
  } else {                                         // W_ih [6][400,300], KT=13 NT=19
    int tt = bid - 1103; int n = tt/247; int rr = tt - n*247; nt = rr/13; kt = rr - nt*13;
    src = Wih + (size_t)n*400*300; srcK = 400; srcC = 300; ldc = 300;
    dst = (__bf16*)(ws + OFF_WIH) + (size_t)tt*512;
  }
  const int kb = kt*32 + ((l & 16) ? 8 : 0);
  const int c  = nt*16 + (l & 15);
  #pragma unroll
  for (int e = 0; e < 16; ++e) {
    int k = kb + e + ((e >= 8) ? 8 : 0);
    float v = (k < srcK && c < srcC) ? src[(size_t)k*ldc + c] : 0.0f;
    dst[l*16 + e] = (__bf16)v;
  }
}

// ---------------------------------------------------------------------------
// Kernel 2: zero h_f32 and h_bf16 (padded cols 100..127 must stay zero).
// ---------------------------------------------------------------------------
__global__ void rims_zero_state(char* __restrict__ ws) {
  unsigned* hf = (unsigned*)(ws + OFF_HF);
  unsigned* hb = (unsigned*)(ws + OFF_HBF);
  const size_t n1 = (size_t)BATCH*6*100;      // f32 words
  const size_t n2 = (size_t)6*256*128/2;      // bf16 pairs as u32
  size_t i = (size_t)blockIdx.x*blockDim.x + threadIdx.x;
  size_t stride = (size_t)gridDim.x*blockDim.x;
  for (size_t j = i; j < n1; j += stride) hf[j] = 0u;
  for (size_t j = i; j < n2; j += stride) hb[j] = 0u;
}

// ---------------------------------------------------------------------------
// Kernel 3: KX = x@Wk (f32), VX = x@Wv (bf16). grid (2048 M-tiles, 6 groups):
// group 0 -> 4 KX col-tiles, groups 1..5 -> 5 VX col-tiles each (25 = 5x5,
// perfectly regular). A (x, f32) loaded once per kt, reused across the group.
// K tail (600 = 18*32+24): kt=18 upper chunk is valid only for lanes<16 ->
// branch-free via dummy address + 0/1 lane mask.
// ---------------------------------------------------------------------------
__global__ void rims_xproj(const float* __restrict__ x, char* __restrict__ ws) {
  const int mt = blockIdx.x;
  const int g  = blockIdx.y;                  // 0 -> KX, 1..5 -> VX
  const int l = threadIdx.x, col = l & 15, hi = l >> 4;
  const int r = mt*16 + col;
  const bool isKX = (g == 0);
  const int NTILES = isKX ? 4 : 5;
  const int ct0 = isKX ? 0 : (g - 1)*5;
  const __bf16* wswb = isKX ? (const __bf16*)(ws + OFF_WK) + (size_t)(ct0*19)*512
                            : (const __bf16*)(ws + OFF_WV) + (size_t)(ct0*19)*512;
  const float* xrow = x + (size_t)r*600;
  v8f acc[5];
  #pragma unroll
  for (int q = 0; q < 5; ++q) acc[q] = (v8f){};

  for (int kt = 0; kt < 19; ++kt) {
    int kb = kt*32 + hi*8;
    union { v16bf v; __bf16 e[16]; } a;
    const float* ap0 = xrow + kb;
    #pragma unroll
    for (int i2 = 0; i2 < 8; ++i2) a.e[i2] = (__bf16)ap0[i2];
    if (kt < 18) {
      #pragma unroll
      for (int i2 = 0; i2 < 8; ++i2) a.e[8+i2] = (__bf16)ap0[16+i2];
    } else {
      // 600 = 18*32+24: upper chunk valid only for hi==0; dummy addr + mask.
      const float* ap1 = xrow + (hi ? 584 : 592);
      float zm = hi ? 0.0f : 1.0f;
      #pragma unroll
      for (int i2 = 0; i2 < 8; ++i2) a.e[8+i2] = (__bf16)(ap1[i2]*zm);
    }
    #pragma unroll
    for (int q = 0; q < 5; ++q) {
      if (q < NTILES) {
        v16bf b = *(const v16bf*)(wswb + ((size_t)q*19 + kt)*512 + l*16);
        acc[q] = wmma_bf16(a.v, b, acc[q]);
      }
    }
  }
  if (isKX) {
    float* KXp = (float*)(ws + OFF_KX);
    #pragma unroll
    for (int q = 0; q < 4; ++q) {
      int cb = q*16 + col;
      #pragma unroll
      for (int v = 0; v < 8; ++v)
        KXp[(size_t)(mt*16 + v + hi*8)*64 + cb] = acc[q][v];
    }
  } else {
    __bf16* VXp = (__bf16*)(ws + OFF_VX);
    #pragma unroll
    for (int q = 0; q < 5; ++q) {
      int cb = (ct0 + q)*16 + col;
      #pragma unroll
      for (int v = 0; v < 8; ++v)
        VXp[(size_t)(mt*16 + v + hi*8)*400 + cb] = (__bf16)acc[q][v];
    }
  }
}

// ---------------------------------------------------------------------------
// Kernel 4: UX[tb,n,:] = VX[tb,:] @ W_ih[n] (bf16 out, f32 acc).
// Register-blocked 2 M-tiles x 5 N-tiles per wave (80 acc VGPRs).
// grid (1024 M-pairs, 6 rims * 4 col-groups). ct = cg*5+q, guard ct<19.
// K tail (400 = 12*32+16): kt=12 upper chunk invalid for ALL lanes -> uniform 0.
// ---------------------------------------------------------------------------
__global__ void rims_uproj(char* __restrict__ ws) {
  const int mt = blockIdx.x;                  // 0..1023, 32 rows each
  const int gj = blockIdx.y;                  // 0..23
  const int n = gj >> 2, cg = gj & 3;
  const int ct0 = cg*5;
  const int l = threadIdx.x, col = l & 15, hi = l >> 4;
  const __bf16* VXp = (const __bf16*)(ws + OFF_VX);
  const __bf16* WIHsw = (const __bf16*)(ws + OFF_WIH);
  const __bf16* row0 = VXp + (size_t)(mt*32 + col)*400;
  const __bf16* row1 = VXp + (size_t)(mt*32 + 16 + col)*400;
  v8f acc[2][5];
  #pragma unroll
  for (int m = 0; m < 2; ++m)
    #pragma unroll
    for (int q = 0; q < 5; ++q) acc[m][q] = (v8f){};

  for (int kt = 0; kt < 13; ++kt) {
    int kb = kt*32 + hi*8;
    union { v16bf v; v8bf h[2]; } a0, a1;
    a0.h[0] = *(const v8bf*)(row0 + kb);
    a1.h[0] = *(const v8bf*)(row1 + kb);
    if (kt < 12) {
      a0.h[1] = *(const v8bf*)(row0 + kb + 16);
      a1.h[1] = *(const v8bf*)(row1 + kb + 16);
    } else {
      a0.h[1] = bz8();
      a1.h[1] = bz8();
    }
    #pragma unroll
    for (int q = 0; q < 5; ++q) {
      int ctq = ct0 + q;
      int ctc = (ctq < 19) ? ctq : 18;        // clamp: dummy work, discarded
      v16bf b = *(const v16bf*)(WIHsw + ((size_t)n*247 + ctc*13 + kt)*512 + l*16);
      acc[0][q] = wmma_bf16(a0.v, b, acc[0][q]);
      acc[1][q] = wmma_bf16(a1.v, b, acc[1][q]);
    }
  }
  __bf16* UXp = (__bf16*)(ws + OFF_UX);
  #pragma unroll
  for (int q = 0; q < 5; ++q) {
    int ctq = ct0 + q;
    if (ctq < 19) {
      int cb = ctq*16 + col;
      #pragma unroll
      for (int m = 0; m < 2; ++m)
        #pragma unroll
        for (int v = 0; v < 8; ++v)
          UXp[((size_t)(mt*32 + m*16 + v + hi*8)*6 + n)*304 + cb] = (__bf16)acc[m][q][v];
    }
  }
}

// ---------------------------------------------------------------------------
// Kernel 5: full recurrence, one launch. 16 WGs x 512 threads (16 waves).
// Per step (3 barriers):
//   A) 24 q-jobs: 4 WMMA, multiply accumulator by kx, shfl_xor-reduce over the
//      16 column lanes, write row sums to s_part[6][4][16].  (+ UX prefetch)
//   B) lanes 0..15 of wave 0: p = sigmoid(scale*s), top-4 mask; other waves
//      immediately start C (disjoint LDS).
//   C) 114 gh-jobs: gh[n] = h@W_hh[n] staged bf16 in LDS for ALL rims.
//   D) elementwise GRU over 6*16*100 elems; writes out, h_f32, h_bf16.
// ---------------------------------------------------------------------------
__global__ void __launch_bounds__(512) rims_recurrence(
    char* __restrict__ ws, const float* __restrict__ b_ih,
    const float* __restrict__ b_hh, float* __restrict__ out) {
  __shared__ __bf16 gh_s[N_RIMS][16][304];   // 58368 B
  __shared__ float  s_part[N_RIMS][4][16];   // 1536 B
  __shared__ float  p_s[16][N_RIMS];
  __shared__ float  m_s[16][N_RIMS];

  const int tid = threadIdx.x;
  const int w = tid >> 5, l = tid & 31, col = l & 15, hi = l >> 4;
  const int r0 = blockIdx.x * 16;

  const __bf16* HBF  = (const __bf16*)(ws + OFF_HBF);
  __bf16*       HBFw = (__bf16*)(ws + OFF_HBF);
  float*        HF   = (float*)(ws + OFF_HF);
  const float*  KXp  = (const float*)(ws + OFF_KX);
  const __bf16* UXp  = (const __bf16*)(ws + OFF_UX);
  const __bf16* WQsw  = (const __bf16*)(ws + OFF_WQ);
  const __bf16* WHHsw = (const __bf16*)(ws + OFF_WHH);
  const float scale = 0.125f;  // 1/sqrt(D_K=64)

  for (int t = 0; t < T_STEPS; ++t) {
    // Prefetch next step's UX slice (456 x 128B lines).
    if (t + 1 < T_STEPS) {
      const char* uxn = (const char*)(UXp + ((size_t)((t+1)*BATCH + r0)*6)*304);
      for (int i = tid; i < 456; i += 512) __builtin_prefetch(uxn + (size_t)i*128, 0, 0);
    }
    // ---- Phase A: q = h@Wq, fused dot with kx, cross-lane reduce.
    for (int job = w; job < 24; job += 16) {
      int n = job >> 2, ct = job & 3;
      const __bf16* hrow = HBF + (size_t)(n*BATCH + r0 + col)*128;
      const __bf16* wsw = WQsw + ((size_t)n*16 + ct*4)*512;
      v8f acc = {};
      #pragma unroll
      for (int kt = 0; kt < 4; ++kt) {
        v16bf b = *(const v16bf*)(wsw + (size_t)kt*512 + l*16);
        acc = wmma_bf16(load_a_bf16(hrow, kt*32 + hi*8), b, acc);
      }
      // partial[v] = q[row_v, col] * kx[row_v, col]; reduce over 16 col lanes.
      float part[8];
      #pragma unroll
      for (int v = 0; v < 8; ++v) {
        float kxv = KXp[(size_t)(t*BATCH + r0 + v + hi*8)*64 + ct*16 + col];
        part[v] = acc[v] * kxv;
      }
      #pragma unroll
      for (int v = 0; v < 8; ++v) {
        #pragma unroll
        for (int m2 = 1; m2 <= 8; m2 <<= 1)
          part[v] += __shfl_xor(part[v], m2, 32);
      }
      if (col == 0) {
        #pragma unroll
        for (int v = 0; v < 8; ++v) s_part[n][ct][v + hi*8] = part[v];
      }
    }
    __syncthreads();
    // ---- Phase B (wave 0, lanes 0..15): p = sigmoid(scale*s); top-4 mask.
    if (tid < 16) {
      int rr = tid;
      float pl[N_RIMS];
      #pragma unroll
      for (int n = 0; n < N_RIMS; ++n) {
        float s = s_part[n][0][rr] + s_part[n][1][rr] + s_part[n][2][rr] + s_part[n][3][rr];
        pl[n] = 1.0f / (1.0f + __expf(-s * scale));
        p_s[rr][n] = pl[n];
      }
      #pragma unroll
      for (int n = 0; n < N_RIMS; ++n) {
        int cnt = 0;
        #pragma unroll
        for (int m2 = 0; m2 < N_RIMS; ++m2)
          cnt += (pl[m2] > pl[n]) || (pl[m2] == pl[n] && m2 < n);
        m_s[rr][n] = (cnt < K_ACT) ? 1.0f : 0.0f;   // ties -> lower idx (lax.top_k)
      }
    }
    // ---- Phase C: gh = h@W_hh for all rims (runs concurrently with B).
    for (int jj = w; jj < 114; jj += 16) {
      int n = jj/19, ct = jj - n*19;
      const __bf16* hrow = HBF + (size_t)(n*BATCH + r0 + col)*128;
      const __bf16* wsw = WHHsw + ((size_t)n*76 + ct*4)*512;
      v8f acc = {};
      #pragma unroll
      for (int kt = 0; kt < 4; ++kt) {
        v16bf b = *(const v16bf*)(wsw + (size_t)kt*512 + l*16);
        acc = wmma_bf16(load_a_bf16(hrow, kt*32 + hi*8), b, acc);
      }
      #pragma unroll
      for (int v = 0; v < 8; ++v)
        gh_s[n][v + hi*8][ct*16 + col] = (__bf16)acc[v];
    }
    __syncthreads();
    // ---- Phase D: elementwise GRU, all rims.
    for (int n = 0; n < N_RIMS; ++n) {
      const float* bi = b_ih + n*G3;
      const float* bh = b_hh + n*G3;
      for (int e = tid; e < 16*D_HID; e += 512) {
        int rr = e / D_HID, j = e - rr*D_HID;
        int b = r0 + rr;
        float p = p_s[rr][n], m = m_s[rr][n];
        const __bf16* ux = UXp + ((size_t)(t*BATCH + b)*6 + n)*304;
        float gir = p*(float)ux[j]       + bi[j];
        float giz = p*(float)ux[j + 100] + bi[j + 100];
        float gin = p*(float)ux[j + 200] + bi[j + 200];
        float ghr = (float)gh_s[n][rr][j]       + bh[j];
        float ghz = (float)gh_s[n][rr][j + 100] + bh[j + 100];
        float ghn = (float)gh_s[n][rr][j + 200] + bh[j + 200];
        float rg = 1.0f / (1.0f + __expf(-(gir + ghr)));
        float zg = 1.0f / (1.0f + __expf(-(giz + ghz)));
        float ng = tanhf(gin + rg*ghn);
        size_t hidx = ((size_t)b*N_RIMS + n)*D_HID + j;
        float hold  = HF[hidx];
        float hnew  = (1.0f - zg)*ng + zg*hold;
        float hnext = m*hnew + (1.0f - m)*hold;
        out[((size_t)(t*BATCH + b)*N_RIMS + n)*D_HID + j] = m*hnew;
        HF[hidx] = hnext;
        HBFw[(size_t)(n*BATCH + b)*128 + j] = (__bf16)hnext;
      }
    }
    __syncthreads();
  }
}

// ---------------------------------------------------------------------------
extern "C" void kernel_launch(void* const* d_in, const int* in_sizes, int n_in,
                              void* d_out, int out_size, void* d_ws, size_t ws_size,
                              hipStream_t stream) {
  (void)in_sizes; (void)n_in; (void)out_size; (void)ws_size;  // needs ~151 MiB ws
  const float* x   = (const float*)d_in[0];
  const float* Wq  = (const float*)d_in[1];
  const float* Wk  = (const float*)d_in[2];
  const float* Wv  = (const float*)d_in[3];
  const float* Wih = (const float*)d_in[4];
  const float* Whh = (const float*)d_in[5];
  const float* bih = (const float*)d_in[6];
  const float* bhh = (const float*)d_in[7];
  char* ws = (char*)d_ws;

  rims_prep_weights<<<2585, 32, 0, stream>>>(Wq, Wk, Wv, Wih, Whh, ws);
  rims_zero_state<<<256, 256, 0, stream>>>(ws);
  rims_xproj<<<dim3(2048, 6), 32, 0, stream>>>(x, ws);
  rims_uproj<<<dim3(1024, 24), 32, 0, stream>>>(ws);
  rims_recurrence<<<16, 512, 0, stream>>>(ws, bih, bhh, (float*)d_out);
}
